// Block_5342939316551
// MI455X (gfx1250) — matrix operand order repaired and checked by reading, not compiled, not run on hardware
//
#include <hip/hip_runtime.h>
#include <cstdint>

// ---------------- constants from the reference ----------------
#define BB 2
#define TT 2048
#define DD 2048
#define NN 8
#define HH 256
#define FF 16384
#define BT (BB*TT)          // 4096
#define QCOLS (NN*HH)       // 2048
#define BIG_NEG -2.3819763e38f

typedef __attribute__((ext_vector_type(16))) __bf16 v16bf;
typedef __attribute__((ext_vector_type(8)))  float  v8f;
typedef __attribute__((address_space(3))) __bf16 lds_bf16;

// ---- gfx1250 async global->LDS copy (ASYNCcnt-tracked), per-lane b128 ----
__device__ __forceinline__ void async_ld_b128(uint32_t lds_off, uint64_t gaddr) {
    asm volatile("global_load_async_to_lds_b128 %0, %1, off"
                 :: "v"(lds_off), "v"(gaddr) : "memory");
}
__device__ __forceinline__ void wait_async0() {
    asm volatile("s_wait_asynccnt 0x0" ::: "memory");
}

// ================================================================
// elementwise converters
// ================================================================
__global__ void k_conv(const float* __restrict__ in, __bf16* __restrict__ out, long n) {
    long i = (long)blockIdx.x * blockDim.x + threadIdx.x;
    if (i < n) out[i] = (__bf16)in[i];
}

// in[g][d][h] -> out[d][g*Hh + h]
__global__ void k_permw(const float* __restrict__ in, __bf16* __restrict__ out,
                        int G, int D, int Hh) {
    long i = (long)blockIdx.x * blockDim.x + threadIdx.x;
    long total = (long)G * D * Hh;
    if (i >= total) return;
    int h = (int)(i % Hh);
    int d = (int)((i / Hh) % D);
    int g = (int)(i / ((long)D * Hh));
    out[(long)d * (G * Hh) + g * Hh + h] = (__bf16)in[i];
}

// strided copy+convert: out[r*ldo + c] = in[r*ldi + off + c]
__global__ void k_cconv(const float* __restrict__ in, __bf16* __restrict__ out,
                        int rows, int cols, int ldi, int off, int ldo) {
    long i = (long)blockIdx.x * blockDim.x + threadIdx.x;
    long total = (long)rows * cols;
    if (i >= total) return;
    int c = (int)(i % cols);
    long r = i / cols;
    out[r * ldo + c] = (__bf16)in[r * ldi + off + c];
}

// ================================================================
// RMSNorm (fp32 in -> bf16 out), one block per row
// ================================================================
__global__ __launch_bounds__(256)
void k_rmsnorm(const float* __restrict__ x, const float* __restrict__ scale,
               __bf16* __restrict__ out, int D) {
    __shared__ float red[256];
    long row = blockIdx.x;
    const float* xr = x + row * D;
    float ss = 0.f;
    for (int c = threadIdx.x; c < D; c += 256) { float v = xr[c]; ss += v * v; }
    red[threadIdx.x] = ss;
    __syncthreads();
    for (int off = 128; off; off >>= 1) {
        if ((int)threadIdx.x < off) red[threadIdx.x] += red[threadIdx.x + off];
        __syncthreads();
    }
    float r = rsqrtf(red[0] / D + 1e-6f);
    for (int c = threadIdx.x; c < D; c += 256)
        out[row * D + c] = (__bf16)(xr[c] * r * (1.f + scale[c]));
}

// ================================================================
// RoPE: in fp32 (row stride ldi), out bf16 (row stride ldo), nh heads of 256
// ================================================================
__global__ void k_rope(const float* __restrict__ in, const int* __restrict__ pos,
                       __bf16* __restrict__ out, int BTr, int nh, int ldi, int ldo,
                       float oscale) {
    const int half = HH / 2;
    long i = (long)blockIdx.x * blockDim.x + threadIdx.x;
    long total = (long)BTr * nh * half;
    if (i >= total) return;
    int j  = (int)(i % half);
    int n  = (int)((i / half) % nh);
    long bt = i / ((long)half * nh);
    float p  = (float)pos[bt];
    float ts = powf(10000.f, (2.f / (float)HH) * (float)j);
    float rad = p / ts;
    float sn = sinf(rad), cs = cosf(rad);
    long ib = bt * ldi + (long)n * HH;
    long ob = bt * ldo + (long)n * HH;
    float x1 = in[ib + j], x2 = in[ib + j + half];
    out[ob + j]        = (__bf16)((x1 * cs - x2 * sn) * oscale);
    out[ob + j + half] = (__bf16)((x2 * cs + x1 * sn) * oscale);
}

// ================================================================
// GELU(tanh) * up  (bf16 in/out)
// ================================================================
__global__ void k_gelu_mul(const __bf16* __restrict__ g, const __bf16* __restrict__ u,
                           __bf16* __restrict__ out, long n) {
    long i = (long)blockIdx.x * blockDim.x + threadIdx.x;
    if (i >= n) return;
    float x = (float)g[i];
    float t = tanhf(0.7978845608f * (x + 0.044715f * x * x * x));
    out[i] = (__bf16)(0.5f * x * (1.f + t) * (float)u[i]);
}

// ================================================================
// bf16 WMMA GEMM: C[M,N] = A[M,K] * B[K,N]
// block tile 64x128, 8 waves (2x4), each wave -> 32x32 (2x2 WMMA C tiles)
// A staged row-major via async global->LDS b128; B staged transposed so
// both operand fragments are contiguous ds_load_b128 pairs.
// EPI: 0 = store f32, 1 = store bf16, 2 = f32 out = acc + res
// ================================================================
template<int EPI>
__global__ __launch_bounds__(256)
void k_gemm_bf16(const __bf16* __restrict__ A, const __bf16* __restrict__ B,
                 const float* __restrict__ res, void* __restrict__ outp,
                 int M, int N, int K) {
    __shared__ __bf16 sA [64 * 32];       // [m][k]
    __shared__ __bf16 sBt[128 * 32];      // [n][k]  (transposed)
    const int tid  = threadIdx.x;
    const int lane = tid & 31;
    const int wave = tid >> 5;            // 0..7
    const int bm = blockIdx.y * 64;
    const int bn = blockIdx.x * 128;
    const int m0 = (wave >> 2) * 32;      // 0,32
    const int n0 = (wave & 3) * 32;       // 0,32,64,96
    const uint32_t sA_base = (uint32_t)(uintptr_t)(lds_bf16*)&sA[0];

    v8f acc00 = {}, acc01 = {}, acc10 = {}, acc11 = {};

    // per-thread staging coordinates
    const int ar = tid >> 2;              // A row 0..63
    const int ac = (tid & 3) * 8;         // A col chunk

    for (int kk = 0; kk < K; kk += 32) {
        // ---- A tile 64x32 via async b128 (one transfer per thread) ----
        async_ld_b128(sA_base + (uint32_t)(ar * 32 + ac) * 2,
                      (uint64_t)(uintptr_t)&A[(long)(bm + ar) * K + kk + ac]);
        // ---- B tile 32x128, stored transposed ----
        for (int j = 0; j < 2; ++j) {
            int idx = j * 256 + tid;          // 512 chunks of 8 halves
            int kr  = idx >> 4;               // 0..31
            int cc  = (idx & 15) * 8;         // 0..120
            uint4 bv = *(const uint4*)&B[(long)(kk + kr) * N + bn + cc];
            const __bf16* be = (const __bf16*)&bv;
            for (int e = 0; e < 8; ++e) sBt[(cc + e) * 32 + kr] = be[e];
        }
        // prefetch next K-chunk into cache while we compute
        if (kk + 32 < K) {
            __builtin_prefetch(&A[(long)(bm + ar) * K + kk + 32 + ac], 0, 3);
            __builtin_prefetch(&B[(long)(kk + 32 + (tid >> 4)) * N + bn + (tid & 15) * 8], 0, 3);
        }
        wait_async0();
        __syncthreads();

        // ---- fragments (ISA 16-bit A 16x32 layout; B via transposed tile) ----
        union F { uint4 u[2]; v16bf v; };
        F a0, a1, b0, b1;
        {
            int r  = m0 + (lane & 15);
            int kb = (lane >> 4) * 8;
            a0.u[0] = *(const uint4*)&sA[r * 32 + kb];
            a0.u[1] = *(const uint4*)&sA[r * 32 + kb + 16];
            a1.u[0] = *(const uint4*)&sA[(r + 16) * 32 + kb];
            a1.u[1] = *(const uint4*)&sA[(r + 16) * 32 + kb + 16];
        }
        {
            int n  = n0 + (lane & 15);
            int b2 = (lane >> 4) * 8;
            b0.u[0] = *(const uint4*)&sBt[n * 32 + b2];
            b0.u[1] = *(const uint4*)&sBt[n * 32 + b2 + 16];
            b1.u[0] = *(const uint4*)&sBt[(n + 16) * 32 + b2];
            b1.u[1] = *(const uint4*)&sBt[(n + 16) * 32 + b2 + 16];
        }
        acc00 = __builtin_amdgcn_wmma_f32_16x16x32_bf16(false, a0.v, false, b0.v, (short)0, acc00, false, false);
        acc01 = __builtin_amdgcn_wmma_f32_16x16x32_bf16(false, a0.v, false, b1.v, (short)0, acc01, false, false);
        acc10 = __builtin_amdgcn_wmma_f32_16x16x32_bf16(false, a1.v, false, b0.v, (short)0, acc10, false, false);
        acc11 = __builtin_amdgcn_wmma_f32_16x16x32_bf16(false, a1.v, false, b1.v, (short)0, acc11, false, false);
        __syncthreads();
    }
    // C layout: row = base + v + 8*(lane>=16), col = base + lane&15
    int col0  = bn + n0 + (lane & 15);
    int rbase = bm + m0 + ((lane >> 4) * 8);
    for (int v = 0; v < 8; ++v) {
        long r0 = (long)(rbase + v) * N;
        long r1 = (long)(rbase + 16 + v) * N;
        long i00 = r0 + col0, i01 = r0 + col0 + 16;
        long i10 = r1 + col0, i11 = r1 + col0 + 16;
        if (EPI == 0) {
            ((float*)outp)[i00] = acc00[v]; ((float*)outp)[i01] = acc01[v];
            ((float*)outp)[i10] = acc10[v]; ((float*)outp)[i11] = acc11[v];
        } else if (EPI == 1) {
            ((__bf16*)outp)[i00] = (__bf16)acc00[v]; ((__bf16*)outp)[i01] = (__bf16)acc01[v];
            ((__bf16*)outp)[i10] = (__bf16)acc10[v]; ((__bf16*)outp)[i11] = (__bf16)acc11[v];
        } else {
            ((float*)outp)[i00] = acc00[v] + res[i00]; ((float*)outp)[i01] = acc01[v] + res[i01];
            ((float*)outp)[i10] = acc10[v] + res[i10]; ((float*)outp)[i11] = acc11[v] + res[i11];
        }
    }
}

// ================================================================
// Flash attention, causal, 1 KV head. Block: 128 thr (4 waves),
// each wave owns a 16-row Q tile; block covers 64 q rows of one head.
// Q/K/V are bf16; Q pre-scaled by H^-0.5. Output enc bf16 (BT x N*H).
// ================================================================
__global__ __launch_bounds__(128)
void k_attn(const __bf16* __restrict__ Q, const __bf16* __restrict__ Kk,
            const __bf16* __restrict__ V, __bf16* __restrict__ O) {
    __shared__ __bf16 sK [32 * HH];      // [s][h]
    __shared__ __bf16 sVt[HH * 32];      // [h][s]  (transposed)
    __shared__ __bf16 sP [4 * 16 * 32];  // per-wave P tile
    const int tid  = threadIdx.x;
    const int lane = tid & 31;
    const int wave = tid >> 5;
    const int qblk = blockIdx.x;
    const int head = blockIdx.y;
    const int b    = blockIdx.z;
    const long rowbase = (long)b * TT + qblk * 64;
    const int  qcol    = head * HH;

    // preload Q fragments for this wave's 16-row tile: 8 chunks of K=32
    union { uint4 u[2]; v16bf v; } qa[8];
    {
        long r  = rowbase + wave * 16 + (lane & 15);
        int  kb = (lane >> 4) * 8;
        const __bf16* qp = &Q[r * QCOLS + qcol];
        for (int c = 0; c < 8; ++c) {
            qa[c].u[0] = *(const uint4*)&qp[c * 32 + kb];
            qa[c].u[1] = *(const uint4*)&qp[c * 32 + kb + 16];
        }
    }
    v8f Oa[16];
    {
        v8f z = {};
        for (int i = 0; i < 16; ++i) Oa[i] = z;
    }
    float mr[8], lr[8], alpha[8];
    for (int v = 0; v < 8; ++v) { mr[v] = -3.0e38f; lr[v] = 0.f; }
    const int qrow_lo = qblk * 64 + wave * 16;
    const int nch = 2 * qblk + 2;

    for (int ch = 0; ch < nch; ++ch) {
        const int sc = ch * 32;
        // stage K chunk (row-major) and V chunk (transposed): 32x256 each
        for (int j = 0; j < 8; ++j) {
            int c  = j * 128 + tid;           // 1024 chunks of 8 halves
            int s  = c >> 5;
            int cc = (c & 31) * 8;
            long gr = ((long)b * TT + sc + s) * HH + cc;
            *(uint4*)&sK[s * HH + cc] = *(const uint4*)&Kk[gr];
            uint4 vv = *(const uint4*)&V[gr];
            const __bf16* ve = (const __bf16*)&vv;
            for (int e = 0; e < 8; ++e) sVt[(cc + e) * 32 + s] = ve[e];
        }
        __syncthreads();

        // logits 16x32 = two 16x16 WMMA tiles, K-reduced over H=256
        v8f t0 = {}, t1 = {};
        {
            int sl = lane & 15;
            int b2 = (lane >> 4) * 8;
            for (int c = 0; c < 8; ++c) {
                union { uint4 u[2]; v16bf v; } kb0, kb1;
                kb0.u[0] = *(const uint4*)&sK[sl * HH + c * 32 + b2];
                kb0.u[1] = *(const uint4*)&sK[sl * HH + c * 32 + b2 + 16];
                kb1.u[0] = *(const uint4*)&sK[(sl + 16) * HH + c * 32 + b2];
                kb1.u[1] = *(const uint4*)&sK[(sl + 16) * HH + c * 32 + b2 + 16];
                t0 = __builtin_amdgcn_wmma_f32_16x16x32_bf16(false, qa[c].v, false, kb0.v, (short)0, t0, false, false);
                t1 = __builtin_amdgcn_wmma_f32_16x16x32_bf16(false, qa[c].v, false, kb1.v, (short)0, t1, false, false);
            }
        }
        // online softmax per row (rows of C tile live across lanes 0-15 / 16-31)
        for (int v = 0; v < 8; ++v) {
            int qr = qrow_lo + v + ((lane >> 4) * 8);
            int s0 = sc + (lane & 15);
            float x0 = (s0      <= qr) ? t0[v] : BIG_NEG;
            float x1 = (s0 + 16 <= qr) ? t1[v] : BIG_NEG;
            float rm = fmaxf(x0, x1);
            for (int off = 8; off; off >>= 1) rm = fmaxf(rm, __shfl_xor(rm, off, 16));
            float mnew = fmaxf(mr[v], rm);
            float a  = __expf(mr[v] - mnew);
            float p0 = __expf(x0 - mnew);
            float p1 = __expf(x1 - mnew);
            float rs = p0 + p1;
            for (int off = 8; off; off >>= 1) rs += __shfl_xor(rs, off, 16);
            lr[v] = lr[v] * a + rs;
            mr[v] = mnew; alpha[v] = a;
            int pr = v + ((lane >> 4) * 8);
            sP[wave * 512 + pr * 32 +      (lane & 15)] = (__bf16)p0;
            sP[wave * 512 + pr * 32 + 16 + (lane & 15)] = (__bf16)p1;
        }
        for (int t = 0; t < 16; ++t)
            for (int v = 0; v < 8; ++v) Oa[t][v] *= alpha[v];

        // P (16x32) @ V (32x256): 16 WMMA tiles over h
        {
            int r   = lane & 15;
            int kb2 = (lane >> 4) * 8;
            int b2  = (lane >> 4) * 8;
            union { uint4 u[2]; v16bf v; } pa;
            pa.u[0] = *(const uint4*)&sP[wave * 512 + r * 32 + kb2];
            pa.u[1] = *(const uint4*)&sP[wave * 512 + r * 32 + kb2 + 16];
            for (int ht = 0; ht < 16; ++ht) {
                union { uint4 u[2]; v16bf v; } vb;
                int hh = ht * 16 + (lane & 15);
                vb.u[0] = *(const uint4*)&sVt[hh * 32 + b2];
                vb.u[1] = *(const uint4*)&sVt[hh * 32 + b2 + 16];
                Oa[ht] = __builtin_amdgcn_wmma_f32_16x16x32_bf16(false, pa.v, false, vb.v, (short)0, Oa[ht], false, false);
            }
        }
        __syncthreads();
    }
    // normalize + store enc
    for (int t = 0; t < 16; ++t)
        for (int v = 0; v < 8; ++v) {
            long r = rowbase + wave * 16 + v + ((lane >> 4) * 8);
            O[r * QCOLS + qcol + t * 16 + (lane & 15)] = (__bf16)(Oa[t][v] / lr[v]);
        }
}

// ================================================================
// host launcher
// ================================================================
static inline dim3 ew_grid(long n) { return dim3((unsigned)((n + 255) / 256)); }

extern "C" void kernel_launch(void* const* d_in, const int* in_sizes, int n_in,
                              void* d_out, int out_size, void* d_ws, size_t ws_size,
                              hipStream_t stream) {
    (void)in_sizes; (void)n_in; (void)out_size; (void)ws_size;
    const float* x       = (const float*)d_in[0];
    const int*   pos     = (const int*)  d_in[1];
    const float* q_w     = (const float*)d_in[3];
    const float* kv_w    = (const float*)d_in[4];
    const float* av_w    = (const float*)d_in[5];
    const float* gate_w  = (const float*)d_in[6];
    const float* lin_w   = (const float*)d_in[7];
    const float* s_attn  = (const float*)d_in[8];
    const float* s_ffw   = (const float*)d_in[9];
    float* out = (float*)d_out;

    // workspace carve-out (256B aligned)
    char* base = (char*)d_ws;
    size_t off = 0;
    auto alloc = [&](size_t bytes) -> char* {
        char* p = base + off;
        off = (off + bytes + 255) & ~(size_t)255;
        return p;
    };
    __bf16* wq   = (__bf16*)alloc((size_t)DD * QCOLS * 2);
    __bf16* wkv  = (__bf16*)alloc((size_t)DD * 512 * 2);
    __bf16* wav  = (__bf16*)alloc((size_t)QCOLS * DD * 2);
    __bf16* wg0  = (__bf16*)alloc((size_t)DD * FF * 2);
    __bf16* wg1  = (__bf16*)alloc((size_t)DD * FF * 2);
    __bf16* wl   = (__bf16*)alloc((size_t)FF * DD * 2);
    __bf16* hbf  = (__bf16*)alloc((size_t)BT * DD * 2);
    float*  qf   = (float*) alloc((size_t)BT * QCOLS * 4);
    float*  kvf  = (float*) alloc((size_t)BT * 512 * 4);
    __bf16* qbf  = (__bf16*)alloc((size_t)BT * QCOLS * 2);
    __bf16* kbf  = (__bf16*)alloc((size_t)BT * HH * 2);
    __bf16* vbf  = (__bf16*)alloc((size_t)BT * HH * 2);
    __bf16* encb = (__bf16*)alloc((size_t)BT * QCOLS * 2);
    float*  x1   = (float*) alloc((size_t)BT * DD * 4);
    __bf16* h2bf = (__bf16*)alloc((size_t)BT * DD * 2);
    __bf16* gbf  = (__bf16*)alloc((size_t)BT * FF * 2);
    __bf16* ubf  = (__bf16*)alloc((size_t)BT * FF * 2);
    __bf16* abf  = (__bf16*)alloc((size_t)BT * FF * 2);

    // ---- weight conversion (fp32 -> bf16, with layout fixes) ----
    k_permw<<<ew_grid((long)NN * DD * HH), 256, 0, stream>>>(q_w,  wq,  NN, DD, HH);
    k_permw<<<ew_grid((long)2  * DD * HH), 256, 0, stream>>>(kv_w, wkv, 2,  DD, HH);
    k_conv <<<ew_grid((long)QCOLS * DD), 256, 0, stream>>>(av_w,          wav, (long)QCOLS * DD);
    k_conv <<<ew_grid((long)DD * FF),    256, 0, stream>>>(gate_w,        wg0, (long)DD * FF);
    k_conv <<<ew_grid((long)DD * FF),    256, 0, stream>>>(gate_w + (long)DD * FF, wg1, (long)DD * FF);
    k_conv <<<ew_grid((long)FF * DD),    256, 0, stream>>>(lin_w,         wl,  (long)FF * DD);

    // ---- attention path ----
    k_rmsnorm<<<BT, 256, 0, stream>>>(x, s_attn, hbf, DD);
    k_gemm_bf16<0><<<dim3(QCOLS / 128, BT / 64), 256, 0, stream>>>(hbf, wq,  nullptr, qf,  BT, QCOLS, DD);
    k_gemm_bf16<0><<<dim3(512   / 128, BT / 64), 256, 0, stream>>>(hbf, wkv, nullptr, kvf, BT, 512,   DD);
    k_rope<<<ew_grid((long)BT * NN * (HH / 2)), 256, 0, stream>>>(qf,  pos, qbf, BT, NN, QCOLS, QCOLS, 0.0625f);
    k_rope<<<ew_grid((long)BT * 1  * (HH / 2)), 256, 0, stream>>>(kvf, pos, kbf, BT, 1,  512,   HH,    1.0f);
    k_cconv<<<ew_grid((long)BT * HH), 256, 0, stream>>>(kvf, vbf, BT, HH, 512, HH, HH);
    k_attn<<<dim3(TT / 64, NN, BB), 128, 0, stream>>>(qbf, kbf, vbf, encb);
    k_gemm_bf16<2><<<dim3(DD / 128, BT / 64), 256, 0, stream>>>(encb, wav, x, x1, BT, DD, QCOLS);

    // ---- FFN path ----
    k_rmsnorm<<<BT, 256, 0, stream>>>(x1, s_ffw, h2bf, DD);
    k_gemm_bf16<1><<<dim3(FF / 128, BT / 64), 256, 0, stream>>>(h2bf, wg0, nullptr, gbf, BT, FF, DD);
    k_gemm_bf16<1><<<dim3(FF / 128, BT / 64), 256, 0, stream>>>(h2bf, wg1, nullptr, ubf, BT, FF, DD);
    k_gelu_mul<<<ew_grid((long)BT * FF), 256, 0, stream>>>(gbf, ubf, abf, (long)BT * FF);
    k_gemm_bf16<2><<<dim3(DD / 128, BT / 64), 256, 0, stream>>>(abf, wl, x1, out, BT, DD, FF);
}
